// SOM_47631187312841
// MI455X (gfx1250) — matrix operand order repaired and checked by reading, not compiled, not run on hardware
//
#include <hip/hip_runtime.h>
#include <hip/hip_bf16.h>

typedef __bf16 bf16_t;
typedef __attribute__((ext_vector_type(8)))  bf16_t v8bf;
typedef __attribute__((ext_vector_type(16))) bf16_t v16bf;
typedef __attribute__((ext_vector_type(8)))  float  v8f;

#define BATCH 4096
#define DIM   1024
#define MN    16384
#define WG_M  128
#define WG_N  128
#define KSTEP 32
#define LDSS  40   // padded row stride in bf16 elements (80B, 16B-aligned chunks)

// Combine two 16B LDS chunks into one v16bf fragment.
__device__ __forceinline__ v16bf ld16(const bf16_t* p0, const bf16_t* p1) {
    v8bf a = *(const v8bf*)p0;
    v8bf b = *(const v8bf*)p1;
    return __builtin_shufflevector(a, b, 0,1,2,3,4,5,6,7,8,9,10,11,12,13,14,15);
}

// Monotone float -> uint mapping (total order, works for negatives too).
__device__ __forceinline__ unsigned key_of(float f) {
    unsigned u = __float_as_uint(f);
    return (u & 0x80000000u) ? ~u : (u | 0x80000000u);
}

__global__ void som_init(unsigned long long* bmu) {
    int i = blockIdx.x * 256 + threadIdx.x;
    if (i < BATCH) bmu[i] = ~0ull;
}

__global__ __launch_bounds__(256)
void som_norms(const float* __restrict__ X, const float* __restrict__ W,
               float* __restrict__ x2, float* __restrict__ w2) {
    int gwave = (blockIdx.x * 256 + threadIdx.x) >> 5;
    int lane  = threadIdx.x & 31;
    if (gwave >= BATCH + MN) return;
    const float* src = (gwave < BATCH) ? (X + (size_t)gwave * DIM)
                                       : (W + (size_t)(gwave - BATCH) * DIM);
    float s = 0.f;
    for (int k = lane * 4; k < DIM; k += 128) {
        float4 v = *(const float4*)(src + k);
        s += v.x * v.x + v.y * v.y + v.z * v.z + v.w * v.w;
    }
#pragma unroll
    for (int m = 1; m < 32; m <<= 1) s += __shfl_xor(s, m, 32);
    if (lane == 0) {
        if (gwave < BATCH) x2[gwave] = s; else w2[gwave - BATCH] = s;
    }
}

// PASS 0: GEMM + per-batch-row argmin (packed u64 atomicMin).
// PASS 1: GEMM recompute + sum(d2 * exp(-manh^2/T^2)) into per-WG partials.
template <int PASS>
__global__ __launch_bounds__(256)
void som_gemm(const float* __restrict__ X, const float* __restrict__ W,
              const float* __restrict__ x2, const float* __restrict__ w2,
              unsigned long long* __restrict__ bmu,
              float* __restrict__ partial) {
    __shared__ bf16_t Ahi[WG_M][LDSS];
    __shared__ bf16_t Alo[WG_M][LDSS];
    __shared__ bf16_t Bhi[WG_N][LDSS];
    __shared__ bf16_t Blo[WG_N][LDSS];
    __shared__ float  wsum[8];

    const int t     = threadIdx.x;
    const int lane  = t & 31;
    const int wave  = t >> 5;
    const int half  = lane >> 4;  // 0/1 (wave32 halves in WMMA layouts)
    const int l15   = lane & 15;
    const int waveM = wave & 3;   // 4 waves along M (32 rows each)
    const int waveN = wave >> 2;  // 2 waves along N (64 cols each)

    const int mBase = blockIdx.y * WG_M;
    const int nBase = blockIdx.x * WG_N;

    // Cooperative loader mapping: thread t handles one row, 16 contiguous K.
    const int ldRow = t >> 1;
    const int ldK   = (t & 1) * 16;
    const float* xsrc = X + (size_t)(mBase + ldRow) * DIM + ldK;
    const float* wsrc = W + (size_t)(nBase + ldRow) * DIM + ldK;

    v8f acc[2][4];
#pragma unroll
    for (int i = 0; i < 2; i++)
#pragma unroll
        for (int j = 0; j < 4; j++) acc[i][j] = (v8f)(0.0f);

    // Software-pipelined staging registers: next k-step's global data.
    float4 vx[4], vw[4];
#pragma unroll
    for (int i = 0; i < 4; i++) {
        vx[i] = *(const float4*)(xsrc + 4 * i);
        vw[i] = *(const float4*)(wsrc + 4 * i);
    }

    for (int k0 = 0; k0 < DIM; k0 += KSTEP) {
        __syncthreads();  // previous iter's fragment reads done before overwrite
        // f32 -> bf16 hi/lo split from staged registers into LDS.
#pragma unroll
        for (int i = 0; i < 4; i++) {
            float fx[4] = {vx[i].x, vx[i].y, vx[i].z, vx[i].w};
            float fw[4] = {vw[i].x, vw[i].y, vw[i].z, vw[i].w};
#pragma unroll
            for (int c = 0; c < 4; c++) {
                bf16_t hx = (bf16_t)fx[c];
                bf16_t hw = (bf16_t)fw[c];
                Ahi[ldRow][ldK + 4 * i + c] = hx;
                Alo[ldRow][ldK + 4 * i + c] = (bf16_t)(fx[c] - (float)hx);
                Bhi[ldRow][ldK + 4 * i + c] = hw;
                Blo[ldRow][ldK + 4 * i + c] = (bf16_t)(fw[c] - (float)hw);
            }
        }
        __syncthreads();

        // Prefetch next k-step's globals; they complete while WMMAs run.
        if (k0 + KSTEP < DIM) {
#pragma unroll
            for (int i = 0; i < 4; i++) {
                vx[i] = *(const float4*)(xsrc + k0 + KSTEP + 4 * i);
                vw[i] = *(const float4*)(wsrc + k0 + KSTEP + 4 * i);
            }
        }

        // Fragment loads per documented wave32 layouts.
        // A 16x32 bf16: lane(<16): row l15, K = {kb..kb+7, kb+16..kb+23}, kb = half*8
        v16bf ah[2], al[2], bh[4], bl[4];
#pragma unroll
        for (int mt = 0; mt < 2; mt++) {
            int ar = waveM * 32 + mt * 16 + l15;
            int kb = half * 8;
            ah[mt] = ld16(&Ahi[ar][kb], &Ahi[ar][kb + 16]);
            al[mt] = ld16(&Alo[ar][kb], &Alo[ar][kb + 16]);
        }
        // B 32x16 bf16: lane holds column l15 (= W row), K = {kb..kb+7, kb+16..kb+23}, kb = half*8
#pragma unroll
        for (int nt = 0; nt < 4; nt++) {
            int br = waveN * 64 + nt * 16 + l15;
            int kb = half * 8;
            bh[nt] = ld16(&Bhi[br][kb], &Bhi[br][kb + 16]);
            bl[nt] = ld16(&Blo[br][kb], &Blo[br][kb + 16]);
        }
        // Split-bf16 fp32-accurate product: hi*hi + hi*lo + lo*hi
#pragma unroll
        for (int mt = 0; mt < 2; mt++) {
#pragma unroll
            for (int nt = 0; nt < 4; nt++) {
                acc[mt][nt] = __builtin_amdgcn_wmma_f32_16x16x32_bf16(
                    false, ah[mt], false, bh[nt], (short)0, acc[mt][nt], false, false);
                acc[mt][nt] = __builtin_amdgcn_wmma_f32_16x16x32_bf16(
                    false, ah[mt], false, bl[nt], (short)0, acc[mt][nt], false, false);
                acc[mt][nt] = __builtin_amdgcn_wmma_f32_16x16x32_bf16(
                    false, al[mt], false, bh[nt], (short)0, acc[mt][nt], false, false);
            }
        }
    }

    // Epilogue. C/D layout: value r of lane = D[mt*16 + half*8 + r][l15]
    const int mWave = mBase + waveM * 32;
    const int nWave = nBase + waveN * 64;
    float w2v[4];
#pragma unroll
    for (int nt = 0; nt < 4; nt++) w2v[nt] = w2[nWave + nt * 16 + l15];

    if (PASS == 0) {
#pragma unroll
        for (int mt = 0; mt < 2; mt++) {
#pragma unroll
            for (int r = 0; r < 8; r++) {
                int b = mWave + mt * 16 + half * 8 + r;
                unsigned long long best = ~0ull;
#pragma unroll
                for (int nt = 0; nt < 4; nt++) {
                    int j = nWave + nt * 16 + l15;
                    float keyf = w2v[nt] - 2.0f * acc[mt][nt][r];  // x2[b] constant per row
                    unsigned long long p =
                        ((unsigned long long)key_of(keyf) << 32) | (unsigned)j;
                    best = best < p ? best : p;
                }
                // min across the 16 lanes of this half (masks < 16 stay in-half)
#pragma unroll
                for (int s = 1; s < 16; s <<= 1) {
                    unsigned lo = __shfl_xor((unsigned)best, s, 32);
                    unsigned hi = __shfl_xor((unsigned)(best >> 32), s, 32);
                    unsigned long long o = ((unsigned long long)hi << 32) | lo;
                    best = best < o ? best : o;
                }
                if (l15 == 0) atomicMin(bmu + b, best);
            }
        }
    } else {
        float lsum = 0.0f;
#pragma unroll
        for (int mt = 0; mt < 2; mt++) {
#pragma unroll
            for (int r = 0; r < 8; r++) {
                int b = mWave + mt * 16 + half * 8 + r;
                unsigned bm = (unsigned)(bmu[b] & 0xFFFFFFFFull);
                int bi = (int)(bm >> 7);
                int bj = (int)(bm & 127u);
                float xb = x2[b];
#pragma unroll
                for (int nt = 0; nt < 4; nt++) {
                    int j = nWave + nt * 16 + l15;
                    float d2 = xb + w2v[nt] - 2.0f * acc[mt][nt][r];
                    int di = (j >> 7) - bi; di = di < 0 ? -di : di;
                    int dj = (j & 127) - bj; dj = dj < 0 ? -dj : dj;
                    float mh = (float)((di + dj) * (di + dj));
                    lsum += d2 * __expf(-mh * 1.0e-4f);  // T^2 = 10000
                }
            }
        }
#pragma unroll
        for (int s = 1; s < 32; s <<= 1) lsum += __shfl_xor(lsum, s, 32);
        if (lane == 0) wsum[wave] = lsum;
        __syncthreads();
        if (t == 0) {
            float s = 0.f;
#pragma unroll
            for (int i = 0; i < 8; i++) s += wsum[i];
            partial[blockIdx.y * gridDim.x + blockIdx.x] = s;
        }
    }
}

// Deterministic fixed-order final reduction over 4096 per-WG partials.
__global__ __launch_bounds__(256)
void som_finalize(const float* __restrict__ partial, float* __restrict__ out) {
    __shared__ float sm[256];
    float s = 0.f;
    for (int i = threadIdx.x; i < 4096; i += 256) s += partial[i];
    sm[threadIdx.x] = s;
    __syncthreads();
    for (int st = 128; st > 0; st >>= 1) {
        if ((int)threadIdx.x < st) sm[threadIdx.x] += sm[threadIdx.x + st];
        __syncthreads();
    }
    if (threadIdx.x == 0) out[0] = sm[0] * (1.0f / 128.0f);  // divide by n
}

extern "C" void kernel_launch(void* const* d_in, const int* in_sizes, int n_in,
                              void* d_out, int out_size, void* d_ws, size_t ws_size,
                              hipStream_t stream) {
    (void)in_sizes; (void)n_in; (void)out_size; (void)ws_size;
    const float* X = (const float*)d_in[0];  // [4096, 1024] f32
    const float* W = (const float*)d_in[1];  // [16384, 1024] f32
    float* out = (float*)d_out;

    char* ws = (char*)d_ws;
    unsigned long long* bmu = (unsigned long long*)ws;        // 32 KB
    float* x2      = (float*)(ws + 32768);                    // 16 KB
    float* w2      = (float*)(ws + 49152);                    // 64 KB
    float* partial = (float*)(ws + 114688);                   // 16 KB

    som_init<<<16, 256, 0, stream>>>(bmu);
    som_norms<<<(BATCH + MN) / 8, 256, 0, stream>>>(X, W, x2, w2);

    dim3 grid(MN / WG_N, BATCH / WG_M);  // (128, 32)
    som_gemm<0><<<grid, 256, 0, stream>>>(X, W, x2, w2, bmu, partial);
    som_gemm<1><<<grid, 256, 0, stream>>>(X, W, x2, w2, bmu, partial);
    som_finalize<<<1, 256, 0, stream>>>(partial, out);
}
